// MultiHeadSelfAttention_45715631899382
// MI455X (gfx1250) — compile-verified, matrix operand order checked by reading, not run on hardware
//
#include <hip/hip_runtime.h>

typedef __attribute__((ext_vector_type(16))) __bf16 bf16x16;
typedef __attribute__((ext_vector_type(8)))  __bf16 bf16x8;
typedef __attribute__((ext_vector_type(8)))  float  f32x8;

#define DIMX   512
#define HEADS  8
#define HDIM   64
#define SEQ    4096
#define BATCH  2
#define KBLK   64   // keys per attention iteration
#define VT_LD  72   // 64 keys + 8 pad (bf16 elems) per V^T row

// ---------------------------------------------------------------------------
// Half-wave swap (lane <-> lane^16). Prefer VALU v_permlanex16_b32 over the
// ds_bpermute path __shfl_xor takes (avoids s_wait_dscnt serialization).
// ---------------------------------------------------------------------------
__device__ __forceinline__ float half_swap(float v) {
#if __has_builtin(__builtin_amdgcn_permlanex16)
    int i = __float_as_int(v);
    i = __builtin_amdgcn_permlanex16(i, i, 0x76543210, (int)0xfedcba98, false, false);
    return __int_as_float(i);
#else
    return __shfl_xor(v, 16, 32);
#endif
}

// ---------------------------------------------------------------------------
// A-style 16x32 bf16 fragment load from a row-major matrix.
// Lane L: row = row0+(L&15); halfwords 0..7 = cols col0+(L>>4)*8+j,
// halfwords 8..15 = cols col0+16+(L>>4)*8+j (CDNA5 16-bit A layout).
// ---------------------------------------------------------------------------
__device__ __forceinline__ bf16x16 load_frag_rm(const __bf16* base, int ld, int lane) {
    int r = lane & 15;
    int h = lane >> 4;
    const __bf16* p = base + (size_t)r * ld + h * 8;
    bf16x8 lo = *reinterpret_cast<const bf16x8*>(p);
    bf16x8 hh = *reinterpret_cast<const bf16x8*>(p + 16);
    bf16x16 f;
#pragma unroll
    for (int i = 0; i < 8; ++i) { f[i] = lo[i]; f[8 + i] = hh[i]; }
    return f;
}

// ---------------------------------------------------------------------------
// f32 -> bf16 conversion, 8 elements per thread
// ---------------------------------------------------------------------------
__global__ void cvt_f32_to_bf16(const float* __restrict__ in,
                                __bf16* __restrict__ out, int n8) {
    int i = blockIdx.x * blockDim.x + threadIdx.x;
    if (i >= n8) return;
    const float4* p = reinterpret_cast<const float4*>(in) + (size_t)i * 2;
    float4 a = p[0], b = p[1];
    bf16x8 o;
    o[0] = (__bf16)a.x; o[1] = (__bf16)a.y; o[2] = (__bf16)a.z; o[3] = (__bf16)a.w;
    o[4] = (__bf16)b.x; o[5] = (__bf16)b.y; o[6] = (__bf16)b.z; o[7] = (__bf16)b.w;
    reinterpret_cast<bf16x8*>(out)[i] = o;
}

// ---------------------------------------------------------------------------
// C = A(M x 512, bf16) @ W^T + bias, W row-major (out,in).
// Transposed tiles: D^T = W_tile x_tile^T. Wave: 32 tokens x 64 features
// (8 WMMAs per k-step vs 6 fragment loads). Block = 8 waves = 256 tokens.
// ---------------------------------------------------------------------------
template <bool OUTF32>
__global__ void __launch_bounds__(256)
gemm_bias_kernel(const __bf16* __restrict__ A, const __bf16* __restrict__ W,
                 const float* __restrict__ bias, void* __restrict__ Cout) {
    const int lane = threadIdx.x & 31;
    const int wave = threadIdx.x >> 5;
    const int mt = (blockIdx.x * 8 + wave) * 32;  // token tile base
    const int n0 = blockIdx.y * 64;               // feature tile base

    f32x8 acc[2][4] = {};

    for (int k0 = 0; k0 < DIMX; k0 += 32) {
        bf16x16 xf0 = load_frag_rm(A + (size_t)mt * DIMX + k0,        DIMX, lane);
        bf16x16 xf1 = load_frag_rm(A + (size_t)(mt + 16) * DIMX + k0, DIMX, lane);
#pragma unroll
        for (int t = 0; t < 4; ++t) {
            bf16x16 wf = load_frag_rm(W + (size_t)(n0 + t * 16) * DIMX + k0, DIMX, lane);
            acc[0][t] = __builtin_amdgcn_wmma_f32_16x16x32_bf16(
                false, wf, false, xf0, (short)0, acc[0][t], false, false);
            acc[1][t] = __builtin_amdgcn_wmma_f32_16x16x32_bf16(
                false, wf, false, xf1, (short)0, acc[1][t], false, false);
        }
    }

    const int hi = lane >> 4;
#pragma unroll
    for (int mi = 0; mi < 2; ++mi) {
        const int tok = mt + mi * 16 + (lane & 15);
#pragma unroll
        for (int t = 0; t < 4; ++t) {
            const int nb = n0 + t * 16 + 8 * hi;  // 8 consecutive features
            f32x8 a = acc[mi][t];
#pragma unroll
            for (int i = 0; i < 8; ++i) a[i] += bias[nb + i];
            if (OUTF32) {
                float* o = reinterpret_cast<float*>(Cout) + (size_t)tok * DIMX + nb;
                float4 s0 = {a[0], a[1], a[2], a[3]};
                float4 s1 = {a[4], a[5], a[6], a[7]};
                *reinterpret_cast<float4*>(o)     = s0;
                *reinterpret_cast<float4*>(o + 4) = s1;
            } else {
                __bf16* o = reinterpret_cast<__bf16*>(Cout) + (size_t)tok * DIMX + nb;
                bf16x8 s;
#pragma unroll
                for (int i = 0; i < 8; ++i) s[i] = (__bf16)a[i];
                *reinterpret_cast<bf16x8*>(o) = s;
            }
        }
    }
}

// ---------------------------------------------------------------------------
// Flash attention, transposed formulation, 64-key blocks.
// Grid: (SEQ/128, HEADS, BATCH), block = 256 (8 waves, 16 queries each).
//   S^T tiles (4x 16keys x 16q) = wmma(K rows from LDS, Q rows)
//   lane = query column -> softmax stats are per-lane + one half_swap
//   P A-fragments (keys 0-31, 32-63) built by in-lane repack of exp values
//   O^T (64d x 16q) += wmma(V^T rows from LDS, P)   [8 WMMAs]
// 16 WMMAs per barrier pair; 64 iterations.
// ---------------------------------------------------------------------------
__global__ void __launch_bounds__(256)
attn_kernel(const __bf16* __restrict__ Qb, const __bf16* __restrict__ Kb,
            const __bf16* __restrict__ Vb, __bf16* __restrict__ Ob) {
    __shared__ __align__(16) __bf16 ldsK[KBLK * HDIM];   // 64 keys x 64 d
    __shared__ __align__(16) __bf16 ldsVT[HDIM * VT_LD]; // 64 d x (64+pad) keys

    const int lane = threadIdx.x & 31;
    const int wave = threadIdx.x >> 5;
    const int h = blockIdx.y;
    const int b = blockIdx.z;
    const size_t headBase = (size_t)b * SEQ * DIMX + (size_t)h * HDIM;

    const int q0g = blockIdx.x * 128 + wave * 16;

    bf16x16 qf0 = load_frag_rm(Qb + headBase + (size_t)q0g * DIMX + 0,  DIMX, lane);
    bf16x16 qf1 = load_frag_rm(Qb + headBase + (size_t)q0g * DIMX + 32, DIMX, lane);

    f32x8 o[4] = {};
    float m_run = -1e30f, l_run = 0.f;
    const float scale = 0.125f;  // 1/sqrt(64)

    // cooperative staging indices: two 8-elem chunks per thread per matrix
    const int key0 = threadIdx.x >> 3;        // 0..31
    const int key1 = key0 + 32;               // 32..63
    const int dg   = threadIdx.x & 7;         // 8-elem d group

    for (int kb = 0; kb < SEQ; kb += KBLK) {
        // ---- stage K row-major and V transposed into LDS ----
        const __bf16* kp0 = Kb + headBase + (size_t)(kb + key0) * DIMX + dg * 8;
        const __bf16* kp1 = Kb + headBase + (size_t)(kb + key1) * DIMX + dg * 8;
        const __bf16* vp0 = Vb + headBase + (size_t)(kb + key0) * DIMX + dg * 8;
        const __bf16* vp1 = Vb + headBase + (size_t)(kb + key1) * DIMX + dg * 8;
        bf16x8 k0 = *reinterpret_cast<const bf16x8*>(kp0);
        bf16x8 k1 = *reinterpret_cast<const bf16x8*>(kp1);
        bf16x8 v0 = *reinterpret_cast<const bf16x8*>(vp0);
        bf16x8 v1 = *reinterpret_cast<const bf16x8*>(vp1);
        *reinterpret_cast<bf16x8*>(&ldsK[key0 * HDIM + dg * 8]) = k0;
        *reinterpret_cast<bf16x8*>(&ldsK[key1 * HDIM + dg * 8]) = k1;
#pragma unroll
        for (int i = 0; i < 8; ++i) {
            ldsVT[(dg * 8 + i) * VT_LD + key0] = v0[i];
            ldsVT[(dg * 8 + i) * VT_LD + key1] = v1[i];
        }
        if (kb + KBLK < SEQ) {  // global_prefetch_b8 next block
            __builtin_prefetch(kp0 + KBLK * DIMX, 0, 1);
            __builtin_prefetch(vp0 + KBLK * DIMX, 0, 1);
            __builtin_prefetch(kp1 + KBLK * DIMX, 0, 1);
            __builtin_prefetch(vp1 + KBLK * DIMX, 0, 1);
        }
        __syncthreads();

        // ---- S^T tiles: 4 x (16 keys x 16 q) ----
        f32x8 st[4];
#pragma unroll
        for (int kt = 0; kt < 4; ++kt) {
            bf16x16 a0 = load_frag_rm(&ldsK[(kt * 16) * HDIM + 0],  HDIM, lane);
            bf16x16 a1 = load_frag_rm(&ldsK[(kt * 16) * HDIM + 32], HDIM, lane);
            f32x8 s = {};
            s = __builtin_amdgcn_wmma_f32_16x16x32_bf16(false, a0, false, qf0, (short)0, s, false, false);
            s = __builtin_amdgcn_wmma_f32_16x16x32_bf16(false, a1, false, qf1, (short)0, s, false, false);
            st[kt] = s;
        }

        // ---- online softmax (lane = query column) ----
        float mx = -1e30f;
#pragma unroll
        for (int kt = 0; kt < 4; ++kt)
#pragma unroll
            for (int i = 0; i < 8; ++i) {
                st[kt][i] *= scale;
                mx = fmaxf(mx, st[kt][i]);
            }
        mx = fmaxf(mx, half_swap(mx));
        const float mnew = fmaxf(m_run, mx);
        const float corr = __expf(m_run - mnew);
        float ps = 0.f;
        bf16x16 pb[2];  // P A-fragments: keys 0-31 and 32-63, built in-lane
#pragma unroll
        for (int kt = 0; kt < 4; ++kt)
#pragma unroll
            for (int i = 0; i < 8; ++i) {
                float p = __expf(st[kt][i] - mnew);
                ps += p;
                pb[kt >> 1][(kt & 1) * 8 + i] = (__bf16)p;
            }
        ps += half_swap(ps);
        l_run = l_run * corr + ps;
        m_run = mnew;
#pragma unroll
        for (int t = 0; t < 4; ++t)
#pragma unroll
            for (int i = 0; i < 8; ++i) o[t][i] *= corr;

        // ---- O^T += V^T * P^T  (4 d-tiles x 2 key halves) ----
#pragma unroll
        for (int t = 0; t < 4; ++t) {
            bf16x16 av0 = load_frag_rm(&ldsVT[(t * 16) * VT_LD + 0],  VT_LD, lane);
            bf16x16 av1 = load_frag_rm(&ldsVT[(t * 16) * VT_LD + 32], VT_LD, lane);
            o[t] = __builtin_amdgcn_wmma_f32_16x16x32_bf16(false, av0, false, pb[0], (short)0, o[t], false, false);
            o[t] = __builtin_amdgcn_wmma_f32_16x16x32_bf16(false, av1, false, pb[1], (short)0, o[t], false, false);
        }
        __syncthreads();
    }

    // ---- normalize + store ----
    const float inv = 1.0f / l_run;
    const int hi  = lane >> 4;
    const int tok = q0g + (lane & 15);
#pragma unroll
    for (int t = 0; t < 4; ++t) {
        bf16x8 s;
#pragma unroll
        for (int i = 0; i < 8; ++i) s[i] = (__bf16)(o[t][i] * inv);
        __bf16* op = Ob + headBase + (size_t)tok * DIMX + t * 16 + 8 * hi;
        *reinterpret_cast<bf16x8*>(op) = s;
    }
}

// ---------------------------------------------------------------------------
extern "C" void kernel_launch(void* const* d_in, const int* in_sizes, int n_in,
                              void* d_out, int out_size, void* d_ws, size_t ws_size,
                              hipStream_t stream) {
    const float* x  = (const float*)d_in[0];
    const float* Wq = (const float*)d_in[1];
    const float* bq = (const float*)d_in[2];
    const float* Wk = (const float*)d_in[3];
    const float* bk = (const float*)d_in[4];
    const float* Wv = (const float*)d_in[5];
    const float* bv = (const float*)d_in[6];
    const float* Wo = (const float*)d_in[7];
    const float* bo = (const float*)d_in[8];

    const size_t M  = (size_t)BATCH * SEQ;  // 8192 tokens
    const size_t xN = M * DIMX;
    const size_t wN = (size_t)DIMX * DIMX;

    char* ws = (char*)d_ws;
    __bf16* xbf = (__bf16*)ws; ws += xN * 2;
    __bf16* wq  = (__bf16*)ws; ws += wN * 2;
    __bf16* wk  = (__bf16*)ws; ws += wN * 2;
    __bf16* wv  = (__bf16*)ws; ws += wN * 2;
    __bf16* wo  = (__bf16*)ws; ws += wN * 2;
    __bf16* qb  = (__bf16*)ws; ws += xN * 2;
    __bf16* kbf = (__bf16*)ws; ws += xN * 2;
    __bf16* vbf = (__bf16*)ws; ws += xN * 2;
    __bf16* ab  = (__bf16*)ws; ws += xN * 2;

    cvt_f32_to_bf16<<<(int)((xN / 8 + 255) / 256), 256, 0, stream>>>(x,  xbf, (int)(xN / 8));
    cvt_f32_to_bf16<<<(int)((wN / 8 + 255) / 256), 256, 0, stream>>>(Wq, wq,  (int)(wN / 8));
    cvt_f32_to_bf16<<<(int)((wN / 8 + 255) / 256), 256, 0, stream>>>(Wk, wk,  (int)(wN / 8));
    cvt_f32_to_bf16<<<(int)((wN / 8 + 255) / 256), 256, 0, stream>>>(Wv, wv,  (int)(wN / 8));
    cvt_f32_to_bf16<<<(int)((wN / 8 + 255) / 256), 256, 0, stream>>>(Wo, wo,  (int)(wN / 8));

    dim3 gg((unsigned)(M / 256), DIMX / 64);  // (32, 8)
    gemm_bias_kernel<false><<<gg, 256, 0, stream>>>(xbf, wq, bq, qb);
    gemm_bias_kernel<false><<<gg, 256, 0, stream>>>(xbf, wk, bk, kbf);
    gemm_bias_kernel<false><<<gg, 256, 0, stream>>>(xbf, wv, bv, vbf);

    attn_kernel<<<dim3(SEQ / 128, HEADS, BATCH), 256, 0, stream>>>(qb, kbf, vbf, ab);

    gemm_bias_kernel<true><<<gg, 256, 0, stream>>>(ab, wo, bo, d_out);
}
